// BitMambaBlock_39006892982870
// MI455X (gfx1250) — compile-verified, matrix operand order checked by reading, not run on hardware
//
#include <hip/hip_runtime.h>
#include <stdint.h>

// ---------------- problem dims (fixed by setup_inputs) ----------------
constexpr int BATCH    = 512;
constexpr int D_MODEL  = 4096;
constexpr int N_HEADS  = 64;
constexpr int D_INNER  = 8192;                 // 2 * d_model
constexpr int DIM_PROJ = 2 * D_INNER + 3 * N_HEADS;  // 16576
constexpr size_t N_WIN  = (size_t)DIM_PROJ * D_MODEL; // 67,895,296
constexpr size_t N_WOUT = (size_t)D_MODEL * D_INNER;  // 33,554,432

// ---------------- workspace layout (bytes) ----------------
constexpr size_t WS_SCALARS = 0;            // 8 floats
constexpr size_t WS_INV_SX  = 1024;         // 512 floats
constexpr size_t WS_INV_SY  = 3072;         // 512 floats
constexpr size_t WS_XQ      = 8192;         // 512*4096 int8  (2 MB)
constexpr size_t WS_YQ      = 4194304;      // 512*8192 int8  (4 MB)
constexpr size_t WS_WQIN    = 8388608;      // 67.9 MB int8
constexpr size_t WS_WQOUT   = WS_WQIN + N_WIN;          // 33.6 MB int8
constexpr size_t WS_ZX      = WS_WQOUT + N_WOUT;        // 512*16576 fp32 (34 MB)

typedef __attribute__((ext_vector_type(8))) int v8i;
typedef __attribute__((ext_vector_type(4))) int v4i_;

// ---------------- CDNA5 async global->LDS helpers ----------------
struct __align__(16) B16 { unsigned long long x, y; };

__device__ __forceinline__ void async_copy16(const void* g, void* l) {
#if __has_builtin(__builtin_amdgcn_global_load_async_to_lds_b128)
    typedef __attribute__((address_space(1))) v4i_* as1p;
    typedef __attribute__((address_space(3))) v4i_* as3p;
    // AS1 address == flat address; AS3 address == low 32 bits of flat LDS
    // address (ISA aperture rule).
    __builtin_amdgcn_global_load_async_to_lds_b128(
        (as1p)(uintptr_t)g, (as3p)(uint32_t)(uintptr_t)l, 0, 0);
#else
    *(B16*)l = *(const B16*)g;   // synchronous fallback (still staged via LDS)
#endif
}

__device__ __forceinline__ void wait_async0() {
#if __has_builtin(__builtin_amdgcn_s_wait_asynccnt)
    __builtin_amdgcn_s_wait_asynccnt(0);
#else
    asm volatile("s_wait_asynccnt 0x0" ::: "memory");
#endif
}

// ---------------- block reductions (wave32, 8 waves / block) ----------------
__device__ __forceinline__ float blockReduceSum(float v, float* sm) {
    #pragma unroll
    for (int o = 16; o > 0; o >>= 1) v += __shfl_down(v, o, 32);
    const int lane = threadIdx.x & 31, w = threadIdx.x >> 5;
    if (lane == 0) sm[w] = v;
    __syncthreads();
    if (w == 0) {
        v = (lane < 8) ? sm[lane] : 0.0f;
        #pragma unroll
        for (int o = 4; o > 0; o >>= 1) v += __shfl_down(v, o, 32);
        if (lane == 0) sm[0] = v;
    }
    __syncthreads();
    float r = sm[0];
    __syncthreads();
    return r;
}

__device__ __forceinline__ float blockReduceMax(float v, float* sm) {
    #pragma unroll
    for (int o = 16; o > 0; o >>= 1) v = fmaxf(v, __shfl_down(v, o, 32));
    const int lane = threadIdx.x & 31, w = threadIdx.x >> 5;
    if (lane == 0) sm[w] = v;
    __syncthreads();
    if (w == 0) {
        v = (lane < 8) ? sm[lane] : 0.0f;   // amax >= 0
        #pragma unroll
        for (int o = 4; o > 0; o >>= 1) v = fmaxf(v, __shfl_down(v, o, 32));
        if (lane == 0) sm[0] = v;
    }
    __syncthreads();
    float r = sm[0];
    __syncthreads();
    return r;
}

// ---------------- 1. init scalar accumulators ----------------
__global__ void init_scalars_kernel(float* sc) {
    if (threadIdx.x < 8) sc[threadIdx.x] = 0.0f;
}

// ---------------- 2. sum(|w|) reduction ----------------
__global__ __launch_bounds__(256)
void absmean_kernel(const float* __restrict__ W, size_t n, float* __restrict__ acc) {
    __shared__ float sm[8];
    float s = 0.0f;
    for (size_t i = (size_t)blockIdx.x * 256 + threadIdx.x; i < n; i += (size_t)gridDim.x * 256)
        s += fabsf(W[i]);
    s = blockReduceSum(s, sm);
    if (threadIdx.x == 0) atomicAdd(acc, s);
}

// ---------------- 3. finalize scales ----------------
__global__ void finalize_scales_kernel(float* sc, float cnt_in, float cnt_out) {
    if (threadIdx.x == 0) {
        float m_in  = fmaxf(sc[0] / cnt_in,  1e-5f);
        sc[2] = 1.0f / m_in;  sc[3] = m_in;
        float m_out = fmaxf(sc[1] / cnt_out, 1e-5f);
        sc[4] = 1.0f / m_out; sc[5] = m_out;
    }
}

// ---------------- 4. ternarize weights to int8 ----------------
__global__ __launch_bounds__(256)
void quantw_kernel(const float* __restrict__ W, int8_t* __restrict__ Wq, size_t n,
                   const float* __restrict__ sc_ptr) {
    const float s = *sc_ptr;
    for (size_t i = (size_t)blockIdx.x * 256 + threadIdx.x; i < n; i += (size_t)gridDim.x * 256) {
        float q = rintf(W[i] * s);
        q = fminf(fmaxf(q, -1.0f), 1.0f);
        Wq[i] = (int8_t)q;
    }
}

// ---------------- 5. RMS-norm + int8 quantize activations ----------------
template <int K>
__global__ __launch_bounds__(256)
void quant_x_kernel(const float* __restrict__ X, const float* __restrict__ normw,
                    int8_t* __restrict__ Xq, float* __restrict__ inv_s) {
    __shared__ float sm[8];
    constexpr int T = K / 256;
    const int row = blockIdx.x;
    const float* x = X + (size_t)row * K;
    float v[T];
    float ssq = 0.0f;
    #pragma unroll
    for (int i = 0; i < T; ++i) {
        float t = x[threadIdx.x + i * 256];
        v[i] = t;
        ssq += t * t;
    }
    ssq = blockReduceSum(ssq, sm);
    const float rms = rsqrtf(ssq / (float)K + 1e-6f);
    float amax = 0.0f;
    #pragma unroll
    for (int i = 0; i < T; ++i) {
        v[i] = v[i] * rms * normw[threadIdx.x + i * 256];
        amax = fmaxf(amax, fabsf(v[i]));
    }
    amax = blockReduceMax(amax, sm);
    const float scale = 127.0f / fmaxf(amax, 1e-5f);
    if (threadIdx.x == 0) inv_s[row] = 1.0f / scale;
    #pragma unroll
    for (int i = 0; i < T; ++i) {
        float q = rintf(v[i] * scale);
        q = fminf(fmaxf(q, -128.0f), 127.0f);
        Xq[(size_t)row * K + threadIdx.x + i * 256] = (int8_t)q;
    }
}

// ---------------- 6/8. int8 x ternary GEMM via V_WMMA_I32_16X16X64_IU8 ----------------
// C[M,N] = Aq[M,K] * Wq[N,K]^T, dequant by inv_srow[m] * (*inv_sw).
// Block tile 128M x 64N; 8 waves stacked along M, each wave 16M x 64N with 4
// independent accumulators (covers the 9-slot IU8 WMMA hazard window).
// Shared B tile (64 cols x 64 K-bytes = 4 KB) is double-buffered in LDS,
// filled with GLOBAL_LOAD_ASYNC_TO_LDS_B128 (ASYNCcnt) so the next k-step's
// fill overlaps this k-step's WMMAs. A fragments are register double-buffered
// direct global loads (zero duplication across waves).
__global__ __launch_bounds__(256)
void gemm_q8_kernel(const int8_t* __restrict__ A, const int8_t* __restrict__ W,
                    float* __restrict__ C, int M, int N, int K,
                    const float* __restrict__ inv_srow, const float* __restrict__ inv_sw_ptr) {
    __shared__ __align__(16) int8_t sB[2][64 * 64];   // 2 x 4 KB

    const int tid  = threadIdx.x;
    const int lane = tid & 31;
    const int wave = tid >> 5;        // 8 waves along M
    const int hi   = lane >> 4;       // K-chunk parity selector (8-bit frag layout)
    const int lo   = lane & 15;

    const int blockM = blockIdx.y * 128;
    const int blockN = blockIdx.x * 64;

    const int rowA = blockM + wave * 16 + lo;
    const int8_t* aRow = A + (size_t)rowA * K;

    // async-staging coordinates: thread t copies 16 bytes of col (t/4), chunk (t%4)
    const int scol   = tid >> 2;            // 0..63  (column within tile)
    const int schunk = (tid & 3) * 16;      // byte offset within 64-byte K-slice
    const int8_t* wSrc = W + (size_t)(blockN + scol) * K + schunk;
    const int sDst = scol * 64 + schunk;    // LDS byte offset inside a buffer

    union Frag { v8i v; unsigned long long d[4]; };
    const v8i vzero = {0, 0, 0, 0, 0, 0, 0, 0};
    v8i acc[4];
    #pragma unroll
    for (int n = 0; n < 4; ++n) acc[n] = vzero;

    // -------- prologue: stage k=0 B tile, load k=0 A fragment --------
    async_copy16(wSrc, &sB[0][sDst]);
    Frag aCur, aNxt;
    #pragma unroll
    for (int j = 0; j < 4; ++j)
        aCur.d[j] = *(const unsigned long long*)(aRow + (2 * j + hi) * 8);
    wait_async0();
    __syncthreads();

    int p = 0;
    for (int k0 = 0; k0 < K; k0 += 64) {
        const bool more = (k0 + 64) < K;
        if (more) {
            // kick off next B tile into the other buffer (ASYNCcnt-tracked)
            async_copy16(wSrc + k0 + 64, &sB[p ^ 1][sDst]);
            // register double-buffer next A fragment
            #pragma unroll
            for (int j = 0; j < 4; ++j)
                aNxt.d[j] = *(const unsigned long long*)(aRow + k0 + 64 + (2 * j + hi) * 8);
        }
        // 4 independent WMMAs on the current buffer
        #pragma unroll
        for (int n = 0; n < 4; ++n) {
            Frag b;
            const int cb = (n * 16 + lo) * 64;
            #pragma unroll
            for (int j = 0; j < 4; ++j)
                b.d[j] = *(const unsigned long long*)&sB[p][cb + (2 * j + hi) * 8];
            acc[n] = __builtin_amdgcn_wmma_i32_16x16x64_iu8(true, aCur.v, true, b.v,
                                                            acc[n], false, false);
        }
        wait_async0();       // next tile fully in LDS (no-op on last iter)
        __syncthreads();     // publish to all waves / allow buffer reuse
        aCur = aNxt;
        p ^= 1;
    }

    // -------- epilogue: dequantize and store --------
    const float invw = *inv_sw_ptr;
    #pragma unroll
    for (int n = 0; n < 4; ++n) {
        const int col = blockN + n * 16 + lo;
        #pragma unroll
        for (int g = 0; g < 8; ++g) {
            // C/D layout: VGPR g = row g (lanes 0-15) / row g+8 (lanes 16-31)
            const int m = blockM + wave * 16 + g + 8 * hi;
            C[(size_t)m * N + col] = (float)acc[n][g] * (inv_srow[m] * invw);
        }
    }
}

// ---------------- 7. fused conv + SSM + gate + RMS-norm + quantize y ----------------
__global__ __launch_bounds__(256)
void fused_ssm_kernel(const float* __restrict__ zx,          // [B, DIM_PROJ]
                      const float* __restrict__ conv_state,  // [B, D_INNER, 3]
                      const float* __restrict__ ssm_state,   // [B, D_INNER]
                      const float* __restrict__ conv_w,      // [D_INNER, 4]
                      const float* __restrict__ conv_b,      // [D_INNER]
                      const float* __restrict__ dt_bias,     // [N_HEADS]
                      const float* __restrict__ A_log,       // [N_HEADS]
                      const float* __restrict__ Dp,          // [D_INNER]
                      const float* __restrict__ norm_out_w,  // [D_INNER]
                      float* __restrict__ out_conv_state,    // [B, D_INNER, 3]
                      float* __restrict__ out_h,             // [B, D_INNER]
                      int8_t* __restrict__ Yq,               // [B, D_INNER]
                      float* __restrict__ inv_sy) {
    __shared__ float sm[8];
    constexpr int T = D_INNER / 256;  // 32
    const int b = blockIdx.x;
    const float* row = zx + (size_t)b * DIM_PROJ;

    float yv[T];
    float ssq = 0.0f;
    #pragma unroll
    for (int i = 0; i < T; ++i) {
        const int c = threadIdx.x + i * 256;
        const int h = c >> 7;                                 // head_dim = 128
        const float z   = row[c];
        const float xin = row[D_INNER + c];
        const float Bv  = row[2 * D_INNER + h];
        const float Cv  = row[2 * D_INNER + N_HEADS + h];
        const float dt  = row[2 * D_INNER + 2 * N_HEADS + h];
        const size_t csb = ((size_t)b * D_INNER + c) * 3;
        const float c0 = conv_state[csb + 0];
        const float c1 = conv_state[csb + 1];
        const float c2 = conv_state[csb + 2];
        const float* cw = conv_w + (size_t)c * 4;
        float xc = c0 * cw[0] + c1 * cw[1] + c2 * cw[2] + xin * cw[3] + conv_b[c];
        const float xt = xc / (1.0f + __expf(-xc));           // silu
        const float dtb = dt + dt_bias[h];
        const float dts = (dtb > 20.0f) ? dtb : log1pf(__expf(dtb));  // softplus
        const float Ah  = -__expf(A_log[h]);
        const float decay = __expf(Ah * dts);
        const float hn = ssm_state[(size_t)b * D_INNER + c] * decay + xt * Bv * dts;
        out_h[(size_t)b * D_INNER + c] = hn;
        out_conv_state[csb + 0] = c1;
        out_conv_state[csb + 1] = c2;
        out_conv_state[csb + 2] = xin;
        float y = hn * Cv + xt * Dp[c];
        y = y * (z / (1.0f + __expf(-z)));                    // gate with silu(z)
        yv[i] = y;
        ssq += y * y;
    }
    ssq = blockReduceSum(ssq, sm);
    const float rms = rsqrtf(ssq / (float)D_INNER + 1e-6f);
    float amax = 0.0f;
    #pragma unroll
    for (int i = 0; i < T; ++i) {
        const int c = threadIdx.x + i * 256;
        yv[i] = yv[i] * rms * norm_out_w[c];
        amax = fmaxf(amax, fabsf(yv[i]));
    }
    amax = blockReduceMax(amax, sm);
    const float scale = 127.0f / fmaxf(amax, 1e-5f);
    if (threadIdx.x == 0) inv_sy[b] = 1.0f / scale;
    #pragma unroll
    for (int i = 0; i < T; ++i) {
        const int c = threadIdx.x + i * 256;
        float q = rintf(yv[i] * scale);
        q = fminf(fmaxf(q, -128.0f), 127.0f);
        Yq[(size_t)b * D_INNER + c] = (int8_t)q;
    }
}

// ---------------- launcher ----------------
extern "C" void kernel_launch(void* const* d_in, const int* in_sizes, int n_in,
                              void* d_out, int out_size, void* d_ws, size_t ws_size,
                              hipStream_t stream) {
    const float* u          = (const float*)d_in[0];
    const float* conv_state = (const float*)d_in[1];
    const float* ssm_state  = (const float*)d_in[2];
    const float* w_in       = (const float*)d_in[3];
    const float* norm_in_w  = (const float*)d_in[4];
    const float* w_out      = (const float*)d_in[5];
    const float* norm_out_w = (const float*)d_in[6];
    const float* conv_w     = (const float*)d_in[7];
    const float* conv_b     = (const float*)d_in[8];
    const float* dt_bias    = (const float*)d_in[9];
    const float* A_log      = (const float*)d_in[10];
    const float* Dp         = (const float*)d_in[11];

    char* wsb = (char*)d_ws;
    float*  scalars = (float*)(wsb + WS_SCALARS);
    float*  inv_sx  = (float*)(wsb + WS_INV_SX);
    float*  inv_sy  = (float*)(wsb + WS_INV_SY);
    int8_t* xq      = (int8_t*)(wsb + WS_XQ);
    int8_t* yq      = (int8_t*)(wsb + WS_YQ);
    int8_t* wq_in   = (int8_t*)(wsb + WS_WQIN);
    int8_t* wq_out  = (int8_t*)(wsb + WS_WQOUT);
    float*  zx      = (float*)(wsb + WS_ZX);

    float* out      = (float*)d_out;                               // [512, 4096]
    float* out_conv = out + (size_t)BATCH * D_MODEL;               // [512, 8192, 3]
    float* out_h    = out_conv + (size_t)BATCH * D_INNER * 3;      // [512, 64, 128]

    // 1-3: weight absmean scales
    init_scalars_kernel<<<1, 32, 0, stream>>>(scalars);
    absmean_kernel<<<1024, 256, 0, stream>>>(w_in,  N_WIN,  &scalars[0]);
    absmean_kernel<<<1024, 256, 0, stream>>>(w_out, N_WOUT, &scalars[1]);
    finalize_scales_kernel<<<1, 32, 0, stream>>>(scalars, (float)N_WIN, (float)N_WOUT);

    // 4: ternarize weights (int8 copies both fit in 192 MB L2 for GEMM re-reads)
    quantw_kernel<<<4096, 256, 0, stream>>>(w_in,  wq_in,  N_WIN,  &scalars[2]);
    quantw_kernel<<<2048, 256, 0, stream>>>(w_out, wq_out, N_WOUT, &scalars[4]);

    // 5: quantize u (RMS-norm + per-row int8)
    quant_x_kernel<D_MODEL><<<BATCH, 256, 0, stream>>>(u, norm_in_w, xq, inv_sx);

    // 6: zxbcdt = xq @ wq_in^T  (IU8 WMMA)
    {
        dim3 grid(DIM_PROJ / 64, BATCH / 128);
        gemm_q8_kernel<<<grid, 256, 0, stream>>>(xq, wq_in, zx, BATCH, DIM_PROJ, D_MODEL,
                                                 inv_sx, &scalars[3]);
    }

    // 7: conv + SSM + gate + norm + quantize y; emits new_conv_state and h_new
    fused_ssm_kernel<<<BATCH, 256, 0, stream>>>(zx, conv_state, ssm_state, conv_w, conv_b,
                                                dt_bias, A_log, Dp, norm_out_w,
                                                out_conv, out_h, yq, inv_sy);

    // 8: out = yq @ wq_out^T (IU8 WMMA)
    {
        dim3 grid(D_MODEL / 64, BATCH / 128);
        gemm_q8_kernel<<<grid, 256, 0, stream>>>(yq, wq_out, out, BATCH, D_MODEL, D_INNER,
                                                 inv_sy, &scalars[5]);
    }
}
